// SharedGlobalAttention_34462817583738
// MI455X (gfx1250) — matrix operand order repaired and checked by reading, not compiled, not run on hardware
//
#include <hip/hip_runtime.h>
#include <hip/hip_bf16.h>

// ---------------------------------------------------------------------------
// SharedGlobalAttention for MI455X (gfx1250, wave32, WMMA 16x16x32 f16,
// LDS-staged flash attention with async global->LDS when available)
// ---------------------------------------------------------------------------
#define D_MODEL   1024
#define N_HEADS   4
#define N_KV      2
#define HEAD_DIM  256
#define L_SEQ     4096
#define BATCH     2
#define N_TOK     (BATCH * L_SEQ)          // 8192

#define K_PAD     264                      // 256 + 8 halves (bank-conflict pad)
#define V_PAD     40                       // 32 + 8 halves

typedef __attribute__((ext_vector_type(16))) _Float16 v16h;
typedef __attribute__((ext_vector_type(8)))  _Float16 v8h;
typedef __attribute__((ext_vector_type(8)))  float    v8f;
typedef __attribute__((ext_vector_type(4)))  int      i4v;

#if __has_builtin(__builtin_amdgcn_global_load_async_to_lds_b128)
#define USE_ASYNC_LDS 1
typedef __attribute__((address_space(1))) i4v* as1_i4p;   // global int4*
typedef __attribute__((address_space(3))) i4v* as3_i4p;   // LDS int4*
#endif

__device__ __forceinline__ void wait_asynccnt0() {
#if __has_builtin(__builtin_amdgcn_s_wait_asynccnt)
  __builtin_amdgcn_s_wait_asynccnt(0);
#else
  asm volatile("s_wait_asynccnt 0" ::: "memory");
#endif
}

__device__ __forceinline__ v8f wmma_f16(v16h a, v16h b, v8f c) {
  return __builtin_amdgcn_wmma_f32_16x16x32_f16(false, a, false, b, (short)0, c, false, false);
}

__device__ __forceinline__ v8f zero8() {
  v8f z;
#pragma unroll
  for (int i = 0; i < 8; ++i) z[i] = 0.0f;
  return z;
}

// A-fragment (16x32 f16): lane holds row m=lane&15; elems 0..7 = K[khalf..+7],
// elems 8..15 = K[16+khalf..+7], khalf = (lane<16 ? 0 : 8).
__device__ __forceinline__ v16h load_a_frag(const _Float16* row, int khalf) {
  v8h lo = *(const v8h*)(row + khalf);
  v8h hi = *(const v8h*)(row + 16 + khalf);
  v16h r;
#pragma unroll
  for (int i = 0; i < 8; ++i) { r[i] = lo[i]; r[i + 8] = hi[i]; }
  return r;
}

// B-fragment: 16 contiguous K halves, only 16B alignment guaranteed (LDS rows)
__device__ __forceinline__ v16h load_b16(const _Float16* p) {
  v8h lo = *(const v8h*)p;
  v8h hi = *(const v8h*)(p + 8);
  v16h r;
#pragma unroll
  for (int i = 0; i < 8; ++i) { r[i] = lo[i]; r[i + 8] = hi[i]; }
  return r;
}

// ---------------------------------------------------------------------------
// 1) RMSNorm -> f16
// ---------------------------------------------------------------------------
__global__ void rmsnorm_kernel(const float* __restrict__ x,
                               const float* __restrict__ w,
                               _Float16* __restrict__ h) {
  __shared__ float red[256];
  const int row = blockIdx.x;
  const float* xr = x + (size_t)row * D_MODEL;
  float s = 0.0f;
  for (int i = threadIdx.x; i < D_MODEL; i += 256) { float v = xr[i]; s += v * v; }
  red[threadIdx.x] = s;
  __syncthreads();
  for (int off = 128; off > 0; off >>= 1) {
    if ((int)threadIdx.x < off) red[threadIdx.x] += red[threadIdx.x + off];
    __syncthreads();
  }
  const float inv = rsqrtf(red[0] * (1.0f / D_MODEL) + 1e-6f);
  for (int i = threadIdx.x; i < D_MODEL; i += 256)
    h[(size_t)row * D_MODEL + i] = (_Float16)(xr[i] * inv * w[i]);
}

// ---------------------------------------------------------------------------
// 2) Weight transpose+convert: WT_f16[n][k] = W_f32[k][n]  (W is K x N)
// ---------------------------------------------------------------------------
__global__ void wtrans_kernel(const float* __restrict__ in, _Float16* __restrict__ out,
                              int K, int N) {
  size_t tid = (size_t)blockIdx.x * 256 + threadIdx.x;
  if (tid >= (size_t)K * N) return;
  int k = (int)(tid % K);
  int n = (int)(tid / K);
  out[tid] = (_Float16)in[(size_t)k * N + n];
}

// ---------------------------------------------------------------------------
// 3) WMMA GEMM, 2x2 register blocking: each wave computes a 32x32 block.
//    C_f32[M x N] = A_f16[M x K] * BT_f16[N x K]^T
// ---------------------------------------------------------------------------
__global__ void gemm_wmma(const _Float16* __restrict__ A,
                          const _Float16* __restrict__ BT,
                          float* __restrict__ C, int M, int N, int K) {
  const int wid  = (int)((blockIdx.x * blockDim.x + threadIdx.x) >> 5);
  const int lane = (int)(threadIdx.x & 31);
  const int btn = N >> 5;
  if (wid >= (M >> 5) * btn) return;
  const int bm = wid / btn;
  const int bn = wid % btn;

  const int m16   = lane & 15;
  const int khalf = (lane >> 4) << 3;
  const int kbase = (lane >> 4) << 4;

  const _Float16* arow0 = A  + (size_t)(bm * 32 + m16) * K;
  const _Float16* arow1 = arow0 + (size_t)16 * K;
  const _Float16* brow0 = BT + (size_t)(bn * 32 + m16) * K;
  const _Float16* brow1 = brow0 + (size_t)16 * K;

  v8f c00 = zero8(), c01 = zero8(), c10 = zero8(), c11 = zero8();
  for (int k0 = 0; k0 < K; k0 += 32) {
    v16h a0 = load_a_frag(arow0 + k0, khalf);
    v16h a1 = load_a_frag(arow1 + k0, khalf);
    v16h b0 = load_a_frag(brow0 + k0, khalf);
    v16h b1 = load_a_frag(brow1 + k0, khalf);
    c00 = wmma_f16(a0, b0, c00);
    c01 = wmma_f16(a0, b1, c01);
    c10 = wmma_f16(a1, b0, c10);
    c11 = wmma_f16(a1, b1, c11);
  }
  const int rhalf = (lane >> 4) << 3;
#pragma unroll
  for (int r = 0; r < 8; ++r) {
    const size_t row0 = (size_t)(bm * 32 + rhalf + r);
    const size_t row1 = row0 + 16;
    C[row0 * N + bn * 32 + m16]      = c00[r];
    C[row0 * N + bn * 32 + 16 + m16] = c01[r];
    C[row1 * N + bn * 32 + m16]      = c10[r];
    C[row1 * N + bn * 32 + 16 + m16] = c11[r];
  }
}

// ---------------------------------------------------------------------------
// 4a) L2-normalize Q, scale by exp(log_temperature), repack [B,H,L,hd] f16
// ---------------------------------------------------------------------------
__global__ void norm_q_kernel(const float* __restrict__ Qf,
                              const float* __restrict__ log_temp,
                              _Float16* __restrict__ Qs) {
  const int lane = threadIdx.x & 31, wave = threadIdx.x >> 5;
  const int widx = blockIdx.x * 8 + wave;
  const int t = widx >> 2;
  const int h = widx & 3;
  const int b = t >> 12;
  const int l = t & (L_SEQ - 1);
  const float* src = Qf + (size_t)t * (N_HEADS * HEAD_DIM) + h * HEAD_DIM;
  float vals[8]; float s = 0.0f;
#pragma unroll
  for (int i = 0; i < 8; ++i) { vals[i] = src[lane * 8 + i]; s += vals[i] * vals[i]; }
#pragma unroll
  for (int off = 1; off < 32; off <<= 1) s += __shfl_xor(s, off, 32);
  const float scale = __expf(log_temp[0]) / fmaxf(sqrtf(s), 1e-12f);
  _Float16* dst = Qs + ((size_t)(b * N_HEADS + h) * L_SEQ + l) * HEAD_DIM;
#pragma unroll
  for (int i = 0; i < 8; ++i) dst[lane * 8 + i] = (_Float16)(vals[i] * scale);
}

// 4b) L2-normalize K, repack [B,KV,L,hd] f16
__global__ void norm_k_kernel(const float* __restrict__ Kf, _Float16* __restrict__ Ks) {
  const int lane = threadIdx.x & 31, wave = threadIdx.x >> 5;
  const int widx = blockIdx.x * 8 + wave;
  const int t  = widx >> 1;
  const int kh = widx & 1;
  const int b = t >> 12;
  const int l = t & (L_SEQ - 1);
  const float* src = Kf + (size_t)t * (N_KV * HEAD_DIM) + kh * HEAD_DIM;
  float vals[8]; float s = 0.0f;
#pragma unroll
  for (int i = 0; i < 8; ++i) { vals[i] = src[lane * 8 + i]; s += vals[i] * vals[i]; }
#pragma unroll
  for (int off = 1; off < 32; off <<= 1) s += __shfl_xor(s, off, 32);
  const float scale = 1.0f / fmaxf(sqrtf(s), 1e-12f);
  _Float16* dst = Ks + ((size_t)(b * N_KV + kh) * L_SEQ + l) * HEAD_DIM;
#pragma unroll
  for (int i = 0; i < 8; ++i) dst[lane * 8 + i] = (_Float16)(vals[i] * scale);
}

// 4c) V transpose to [B,KV,hd,L] f16
__global__ void vtrans_kernel(const float* __restrict__ Vf, _Float16* __restrict__ Vt) {
  size_t tid = (size_t)blockIdx.x * 256 + threadIdx.x;
  const int l  = (int)(tid & (L_SEQ - 1));
  size_t rest  = tid >> 12;
  const int e  = (int)(rest & (HEAD_DIM - 1));
  const int kh = (int)((rest >> 8) & (N_KV - 1));
  const int b  = (int)(rest >> 9);
  Vt[tid] = (_Float16)Vf[((size_t)(b * L_SEQ + l)) * (N_KV * HEAD_DIM) + kh * HEAD_DIM + e];
}

// ---------------------------------------------------------------------------
// 5) Flash attention. 8 waves/block share one (b,h); 128 query rows per block.
//    K/V tiles staged in LDS once per block (async global->LDS if available),
//    scores + PV on WMMA, online softmax with cross-lane reductions.
// ---------------------------------------------------------------------------
__global__ void attn_kernel(const _Float16* __restrict__ Qs,
                            const _Float16* __restrict__ Ks,
                            const _Float16* __restrict__ Vt,
                            _Float16* __restrict__ attn_out) {
  __shared__ __align__(16) _Float16 Klds[32][K_PAD];       // kv-row major, padded
  __shared__ __align__(16) _Float16 Vlds[HEAD_DIM][V_PAD]; // d-major, padded
  __shared__ __align__(16) _Float16 pbuf[8][16 * 32];      // per-wave P tile

  const int lane = threadIdx.x & 31;
  const int wave = threadIdx.x >> 5;
  const int bh   = blockIdx.x >> 5;
  const int qblk = blockIdx.x & 31;
  const int b    = bh >> 2;
  const int h    = bh & 3;
  const int kvh  = h >> 1;                    // GQA N_REP = 2
  const int qbase = qblk * 128 + wave * 16;

  const _Float16* Qb = Qs + ((size_t)(b * N_HEADS + h) * L_SEQ) * HEAD_DIM;
  const _Float16* Kb = Ks + ((size_t)(b * N_KV + kvh) * L_SEQ) * HEAD_DIM;
  const _Float16* Vb = Vt + ((size_t)(b * N_KV + kvh) * HEAD_DIM) * L_SEQ;

  const int m16   = lane & 15;
  const int khalf = (lane >> 4) << 3;
  const int kbase = (lane >> 4) << 4;

  // Q fragments (16 rows x head_dim) held in registers
  v16h qf[8];
  {
    const _Float16* qrow = Qb + (size_t)(qbase + m16) * HEAD_DIM;
#pragma unroll
    for (int c = 0; c < 8; ++c) qf[c] = load_b16(qrow + c * 32 + kbase);
  }

  v8f acc[16];
#pragma unroll
  for (int t = 0; t < 16; ++t) acc[t] = zero8();
  float mrow[8], lrow[8];
#pragma unroll
  for (int r = 0; r < 8; ++r) { mrow[r] = -1e30f; lrow[r] = 0.0f; }

  _Float16* pw = &pbuf[wave][0];
  const int rlane0 = qbase + ((lane >> 4) << 3);
  const int njt = qblk * 4 + 4;               // block-uniform causal extent

  for (int j = 0; j < njt; ++j) {
    const int kcol = j << 5;

    // ---- stage K (32 x 256) and V (256 x 32) tiles into LDS ----
#pragma unroll
    for (int u = 0; u < 4; ++u) {
      const int chunk = (int)threadIdx.x + u * 256;        // 1024 x 16B chunks
      const int krow  = chunk >> 5;
      const int kcoff = (chunk & 31) * 8;
      const _Float16* gsrc = Kb + (size_t)(kcol + krow) * HEAD_DIM + kcoff;
      _Float16* ldst = &Klds[krow][kcoff];
#ifdef USE_ASYNC_LDS
      __builtin_amdgcn_global_load_async_to_lds_b128(
          (as1_i4p)(void*)gsrc, (as3_i4p)(void*)ldst, 0, 0);
#else
      *(v8h*)ldst = *(const v8h*)gsrc;
#endif
    }
#pragma unroll
    for (int u = 0; u < 4; ++u) {
      const int chunk = (int)threadIdx.x + u * 256;
      const int d    = chunk >> 2;
      const int koff = (chunk & 3) * 8;
      const _Float16* gsrc = Vb + (size_t)d * L_SEQ + kcol + koff;
      _Float16* ldst = &Vlds[d][koff];
#ifdef USE_ASYNC_LDS
      __builtin_amdgcn_global_load_async_to_lds_b128(
          (as1_i4p)(void*)gsrc, (as3_i4p)(void*)ldst, 0, 0);
#else
      *(v8h*)ldst = *(const v8h*)gsrc;
#endif
    }
#ifdef USE_ASYNC_LDS
    wait_asynccnt0();
#endif
    __syncthreads();

    if (kcol < qbase + 16) {   // wave-uniform causal guard
      // ---- scores: two 16-wide key tiles from LDS ----
      v8f s0 = zero8(), s1 = zero8();
#pragma unroll
      for (int c = 0; c < 8; ++c) {
        v16h b0 = load_b16(&Klds[m16][c * 32 + kbase]);
        s0 = wmma_f16(qf[c], b0, s0);
        v16h b1 = load_b16(&Klds[16 + m16][c * 32 + kbase]);
        s1 = wmma_f16(qf[c], b1, s1);
      }
      // ---- causal mask + online softmax ----
      const int col0 = kcol + m16;
      const int col1 = kcol + 16 + m16;
#pragma unroll
      for (int r = 0; r < 8; ++r) {
        const int row = rlane0 + r;
        float a0 = (col0 <= row) ? s0[r] : -1e30f;
        float a1 = (col1 <= row) ? s1[r] : -1e30f;
        float mx = fmaxf(a0, a1);
#pragma unroll
        for (int off = 1; off < 16; off <<= 1) mx = fmaxf(mx, __shfl_xor(mx, off, 32));
        const float mnew = fmaxf(mrow[r], mx);
        const float p0 = __expf(a0 - mnew);
        const float p1 = __expf(a1 - mnew);
        float ps = p0 + p1;
#pragma unroll
        for (int off = 1; off < 16; off <<= 1) ps += __shfl_xor(ps, off, 32);
        const float fs = __expf(mrow[r] - mnew);
        lrow[r] = lrow[r] * fs + ps;
        mrow[r] = mnew;
#pragma unroll
        for (int t = 0; t < 16; ++t) acc[t][r] *= fs;
        const int M = ((lane >> 4) << 3) + r;
        pw[M * 32 + m16]      = (_Float16)p0;
        pw[M * 32 + 16 + m16] = (_Float16)p1;
      }
      __builtin_amdgcn_wave_barrier();
      asm volatile("s_wait_dscnt 0" ::: "memory");
      // ---- reload P as WMMA A-fragment ----
      v16h pf;
      {
        const _Float16* pp = pw + m16 * 32;
        v8h lo = *(const v8h*)(pp + khalf);
        v8h hi = *(const v8h*)(pp + 16 + khalf);
#pragma unroll
        for (int i = 0; i < 8; ++i) { pf[i] = lo[i]; pf[i + 8] = hi[i]; }
      }
      __builtin_amdgcn_wave_barrier();
      // ---- PV: 16 output d-tiles, B-frag from LDS V tile ----
#pragma unroll
      for (int t = 0; t < 16; ++t) {
        v16h bf = load_a_frag(&Vlds[t * 16 + m16][0], khalf);
        acc[t] = wmma_f16(pf, bf, acc[t]);
      }
    }
    __syncthreads();
  }

  // ---- epilogue ----
  float inv[8];
#pragma unroll
  for (int r = 0; r < 8; ++r) inv[r] = (lrow[r] > 0.0f) ? (1.0f / lrow[r]) : 0.0f;
#pragma unroll
  for (int t = 0; t < 16; ++t) {
#pragma unroll
    for (int r = 0; r < 8; ++r) {
      const int row = rlane0 + r;
      const int d   = t * 16 + m16;
      attn_out[((size_t)(b * L_SEQ + row)) * D_MODEL + h * HEAD_DIM + d] =
          (_Float16)(acc[t][r] * inv[r]);
    }
  }
}

// ---------------------------------------------------------------------------
// 6) Output projection + gated residual (2x2 WMMA blocking):
//    out = x + gate * (attn @ Wo)
// ---------------------------------------------------------------------------
__global__ void gemm_out_wmma(const _Float16* __restrict__ A,
                              const _Float16* __restrict__ BT,
                              const float* __restrict__ x,
                              const float* __restrict__ gatep,
                              float* __restrict__ out, int M, int N, int K) {
  const int wid  = (int)((blockIdx.x * blockDim.x + threadIdx.x) >> 5);
  const int lane = (int)(threadIdx.x & 31);
  const int btn = N >> 5;
  if (wid >= (M >> 5) * btn) return;
  const int bm = wid / btn;
  const int bn = wid % btn;

  const int m16   = lane & 15;
  const int khalf = (lane >> 4) << 3;
  const int kbase = (lane >> 4) << 4;

  const _Float16* arow0 = A  + (size_t)(bm * 32 + m16) * K;
  const _Float16* arow1 = arow0 + (size_t)16 * K;
  const _Float16* brow0 = BT + (size_t)(bn * 32 + m16) * K;
  const _Float16* brow1 = brow0 + (size_t)16 * K;

  v8f c00 = zero8(), c01 = zero8(), c10 = zero8(), c11 = zero8();
  for (int k0 = 0; k0 < K; k0 += 32) {
    v16h a0 = load_a_frag(arow0 + k0, khalf);
    v16h a1 = load_a_frag(arow1 + k0, khalf);
    v16h b0 = load_a_frag(brow0 + k0, khalf);
    v16h b1 = load_a_frag(brow1 + k0, khalf);
    c00 = wmma_f16(a0, b0, c00);
    c01 = wmma_f16(a0, b1, c01);
    c10 = wmma_f16(a1, b0, c10);
    c11 = wmma_f16(a1, b1, c11);
  }
  const float g = gatep[0];
  const int rhalf = (lane >> 4) << 3;
#pragma unroll
  for (int r = 0; r < 8; ++r) {
    const size_t row0 = (size_t)(bm * 32 + rhalf + r);
    const size_t row1 = row0 + 16;
    size_t i00 = row0 * N + bn * 32 + m16;
    size_t i01 = i00 + 16;
    size_t i10 = row1 * N + bn * 32 + m16;
    size_t i11 = i10 + 16;
    out[i00] = x[i00] + g * c00[r];
    out[i01] = x[i01] + g * c01[r];
    out[i10] = x[i10] + g * c10[r];
    out[i11] = x[i11] + g * c11[r];
  }
}

// ---------------------------------------------------------------------------
// Host launcher
// ---------------------------------------------------------------------------
extern "C" void kernel_launch(void* const* d_in, const int* in_sizes, int n_in,
                              void* d_out, int out_size, void* d_ws, size_t ws_size,
                              hipStream_t stream) {
  const float* x        = (const float*)d_in[0];
  const float* norm_w   = (const float*)d_in[1];
  const float* Wq       = (const float*)d_in[2];
  const float* Wk       = (const float*)d_in[3];
  const float* Wv       = (const float*)d_in[4];
  const float* Wo       = (const float*)d_in[5];
  const float* gate     = (const float*)d_in[6];
  const float* log_temp = (const float*)d_in[7];
  float* out = (float*)d_out;

  char* ws = (char*)d_ws;
  size_t off = 0;
  auto take = [&](size_t bytes) { char* p = ws + off; off += (bytes + 255) & ~(size_t)255; return p; };

  _Float16* h16 = (_Float16*)take((size_t)N_TOK * D_MODEL * 2);
  _Float16* WqT = (_Float16*)take((size_t)D_MODEL * D_MODEL * 2);
  _Float16* WkT = (_Float16*)take((size_t)512 * D_MODEL * 2);
  _Float16* WvT = (_Float16*)take((size_t)512 * D_MODEL * 2);
  _Float16* WoT = (_Float16*)take((size_t)D_MODEL * D_MODEL * 2);
  float*    Qf  = (float*)take((size_t)N_TOK * D_MODEL * 4);
  float*    Kf  = (float*)take((size_t)N_TOK * 512 * 4);
  float*    Vf  = (float*)take((size_t)N_TOK * 512 * 4);
  _Float16* Qs  = (_Float16*)take((size_t)N_TOK * D_MODEL * 2);
  _Float16* Ks  = (_Float16*)take((size_t)N_TOK * 512 * 2);
  _Float16* Vt  = (_Float16*)take((size_t)N_TOK * 512 * 2);
  _Float16* attn16 = (_Float16*)Qf;   // reuse Qf region

  rmsnorm_kernel<<<N_TOK, 256, 0, stream>>>(x, norm_w, h16);

  wtrans_kernel<<<(D_MODEL * D_MODEL) / 256, 256, 0, stream>>>(Wq, WqT, D_MODEL, D_MODEL);
  wtrans_kernel<<<(D_MODEL * 512) / 256, 256, 0, stream>>>(Wk, WkT, D_MODEL, 512);
  wtrans_kernel<<<(D_MODEL * 512) / 256, 256, 0, stream>>>(Wv, WvT, D_MODEL, 512);
  wtrans_kernel<<<(D_MODEL * D_MODEL) / 256, 256, 0, stream>>>(Wo, WoT, D_MODEL, D_MODEL);

  gemm_wmma<<<(N_TOK / 32) * (D_MODEL / 32) / 8, 256, 0, stream>>>(h16, WqT, Qf, N_TOK, D_MODEL, D_MODEL);
  gemm_wmma<<<(N_TOK / 32) * (512 / 32) / 8, 256, 0, stream>>>(h16, WkT, Kf, N_TOK, 512, D_MODEL);
  gemm_wmma<<<(N_TOK / 32) * (512 / 32) / 8, 256, 0, stream>>>(h16, WvT, Vf, N_TOK, 512, D_MODEL);

  norm_q_kernel<<<(N_TOK * N_HEADS) / 8, 256, 0, stream>>>(Qf, log_temp, Qs);
  norm_k_kernel<<<(N_TOK * N_KV) / 8, 256, 0, stream>>>(Kf, Ks);
  vtrans_kernel<<<((size_t)N_TOK * 512) / 256, 256, 0, stream>>>(Vf, Vt);

  attn_kernel<<<BATCH * N_HEADS * (L_SEQ / 128), 256, 0, stream>>>(Qs, Ks, Vt, attn16);

  gemm_out_wmma<<<(N_TOK / 32) * (D_MODEL / 32) / 8, 256, 0, stream>>>(
      attn16, WoT, x, gate, out, N_TOK, D_MODEL, D_MODEL);
}
